// VanillaRNN_43447889166512
// MI455X (gfx1250) — matrix operand order repaired and checked by reading, not compiled, not run on hardware
//
#include <hip/hip_runtime.h>

typedef __attribute__((ext_vector_type(16))) _Float16 v16h;
typedef __attribute__((ext_vector_type(8)))  _Float16 v8h;
typedef __attribute__((ext_vector_type(8)))  float    v8f;

#define T_STEPS 256
#define HIDDEN  256
#define B_TILE  16
#define CLASSES 10
#define NTHREADS 256   // 8 waves (wave32)

__device__ __forceinline__ float fast_tanh(float v) {
#if __has_builtin(__builtin_amdgcn_tanhf)
    return __builtin_amdgcn_tanhf(v);      // gfx1250 v_tanh_f32 (TRANS op)
#else
    float e = __expf(2.0f * v);            // v_exp_f32 based fallback
    return 1.0f - 2.0f / (e + 1.0f);
#endif
}

// Per workgroup: batch rows [b0, b0+16). Per step:
//   h_new[16x256] = tanh( h[16x256] @ w_hh[256x256] + x_t (x) w_hx + b_h )
// via v_wmma_f32_16x16x32_f16. Wave w owns n-tiles {2w, 2w+1}.
// w_hh B-fragments are loaded ONCE into 128 VGPRs/lane and stay register-
// resident across all 256 time steps; only h (A fragments) moves through LDS.
__global__ __launch_bounds__(NTHREADS) void rnn_wmma_kernel(
    const float* __restrict__ x,     // [2048, 256] row-major
    const float* __restrict__ w_hx,  // [256]
    const float* __restrict__ w_hh,  // [256, 256] (w_hh[k][n])
    const float* __restrict__ b_h,   // [256]
    const float* __restrict__ w_ph,  // [256, 10]
    const float* __restrict__ b_p,   // [10]
    float* __restrict__ out)         // [2048, 10]
{
    // w_hh staged (one-time) pre-swizzled into B-fragment order:
    // [tile_n][kb][half][n_local][e], k = kb*32 + half*16 + e, n = tile_n*16 + n_local
    __shared__ _Float16 sWB[16 * 8 * 2 * 16 * 16];       // 128 KB
    __shared__ float    sX[T_STEPS * B_TILE];            //  16 KB, [t][m]
    __shared__ _Float16 sH[2][B_TILE * HIDDEN];          // 2 x 8 KB, [m][k]

    const int tid  = threadIdx.x;
    const int lane = tid & 31;
    const int wave = tid >> 5;                 // 0..7
    const int b0   = blockIdx.x * B_TILE;

    // ---- stage x tile transposed: sX[t][m] (global reads coalesced in t) ----
    for (int i = tid; i < B_TILE * T_STEPS; i += NTHREADS) {
        int m = i >> 8;          // 0..15
        int t = i & 255;
        sX[t * B_TILE + m] = x[(size_t)(b0 + m) * T_STEPS + t];
    }

    // ---- stage w_hh swizzled into B-fragment order (f32 -> f16) ----
    for (int d = tid; d < HIDDEN * HIDDEN; d += NTHREADS) {
        int e  = d & 15;
        int nl = (d >> 4) & 15;
        int hf = (d >> 8) & 1;
        int kb = (d >> 9) & 7;
        int tn = (d >> 12) & 15;
        int k  = kb * 32 + hf * 16 + e;
        int n  = tn * 16 + nl;
        sWB[d] = (_Float16)w_hh[k * HIDDEN + n];
    }

    // ---- h0 = 0 ----
    for (int i = tid; i < B_TILE * HIDDEN; i += NTHREADS)
        sH[0][i] = (_Float16)0.0f;

    // ---- per-lane constants ----
    const int half = lane >> 4;     // K-half selector (A,B) / D row-half
    const int nl   = lane & 15;     // A row m / B,D column n_local
    const int tn0  = wave * 2, tn1 = wave * 2 + 1;
    const int n0   = tn0 * 16 + nl, n1 = tn1 * 16 + nl;
    const float wx0 = w_hx[n0], wx1 = w_hx[n1];
    const float bh0 = b_h[n0],  bh1 = b_h[n1];
    const int m_base = half * 8;    // D VGPR r holds row m = r + 8*half

    __syncthreads();

    // ---- hoist all 16 B fragments into registers (loop-invariant) ----
    v16h bfrag[16];
    #pragma unroll
    for (int kb = 0; kb < 8; ++kb) {
        bfrag[kb]     = *(const v16h*)(sWB + (((tn0 * 8 + kb) * 2 + half) * 16 + nl) * 16);
        bfrag[kb + 8] = *(const v16h*)(sWB + (((tn1 * 8 + kb) * 2 + half) * 16 + nl) * 16);
    }

    int cur = 0;
    for (int t = 0; t < T_STEPS; ++t) {
        const _Float16* hc = sH[cur];
        _Float16*       hn = sH[cur ^ 1];
        v8f acc0 = {};
        v8f acc1 = {};
        #pragma unroll
        for (int kb = 0; kb < 8; ++kb) {
            // A fragment: lane row m = nl; two contiguous 8xf16 K-runs.
            const _Float16* arow = hc + nl * HIDDEN + kb * 32 + half * 16;
            v8h alo = *(const v8h*)(arow);        // K = kb*32 + 16*half + 0..7
            v8h ahi = *(const v8h*)(arow + 8);    // K = kb*32 + 16*half + 8..15
            v16h a = __builtin_shufflevector(alo, ahi,
                     0,1,2,3,4,5,6,7,8,9,10,11,12,13,14,15);
            acc0 = __builtin_amdgcn_wmma_f32_16x16x32_f16(
                       false, a, false, bfrag[kb], (short)0, acc0, false, false);
            acc1 = __builtin_amdgcn_wmma_f32_16x16x32_f16(
                       false, a, false, bfrag[kb + 8], (short)0, acc1, false, false);
        }
        // Epilogue: += x_t * w_hx + b_h, tanh, store next-state (f16).
        v8f xv = *(const v8f*)(sX + t * B_TILE + m_base);  // x for rows m_base..m_base+7
        #pragma unroll
        for (int r = 0; r < 8; ++r) {
            int   m  = m_base + r;
            float xm = xv[r];
            float v0 = acc0[r] + xm * wx0 + bh0;
            float v1 = acc1[r] + xm * wx1 + bh1;
            hn[m * HIDDEN + n0] = (_Float16)fast_tanh(v0);
            hn[m * HIDDEN + n1] = (_Float16)fast_tanh(v1);
        }
        __syncthreads();
        cur ^= 1;
    }

    // ---- final projection: out[b][c] = h_final . w_ph[:,c] + b_p[c] ----
    const _Float16* hfin = sH[cur];
    for (int i = tid; i < B_TILE * CLASSES; i += NTHREADS) {
        int m = i / CLASSES, c = i % CLASSES;
        float s = b_p[c];
        #pragma unroll 8
        for (int k = 0; k < HIDDEN; ++k)
            s += (float)hfin[m * HIDDEN + k] * w_ph[k * CLASSES + c];
        out[(size_t)(b0 + m) * CLASSES + c] = s;
    }
}

extern "C" void kernel_launch(void* const* d_in, const int* in_sizes, int n_in,
                              void* d_out, int out_size, void* d_ws, size_t ws_size,
                              hipStream_t stream) {
    (void)in_sizes; (void)n_in; (void)d_ws; (void)ws_size; (void)out_size;
    const float* x    = (const float*)d_in[0];  // [2048,256]
    const float* w_hx = (const float*)d_in[1];  // [1,256]
    const float* w_hh = (const float*)d_in[2];  // [256,256]
    const float* b_h  = (const float*)d_in[3];  // [256]
    const float* w_ph = (const float*)d_in[4];  // [256,10]
    const float* b_p  = (const float*)d_in[5];  // [10]
    float* out = (float*)d_out;                 // [2048,10]

    dim3 grid(2048 / B_TILE);   // 128 workgroups
    dim3 block(NTHREADS);
    rnn_wmma_kernel<<<grid, block, 0, stream>>>(x, w_hx, w_hh, b_h, w_ph, b_p, out);
}
